// ModernNCA_17532056502380
// MI455X (gfx1250) — matrix operand order repaired and checked by reading, not compiled, not run on hardware
//
#include <hip/hip_runtime.h>
#include <hip/hip_bf16.h>
#include <math.h>
#include <stdint.h>

// ---------------------------------------------------------------------------
// ModernNCA eval-path for MI455X (gfx1250, wave32, WMMA f16 16x16x32).
//  - f16 convert/transpose prep, one-hot(+ones) f16 table for aggregation
//  - ping-pong register-pipelined WMMA GEMM encoder, BN fused into epilogues
//  - TDM (tensor_load_to_lds) stages the Q tile for the distance kernels
//  - two-pass flash-softmax distance stage; class bins via WMMA vs YhT
// ---------------------------------------------------------------------------

typedef __attribute__((ext_vector_type(16))) _Float16 v16h;
typedef __attribute__((ext_vector_type(8)))  _Float16 v8h;
typedef __attribute__((ext_vector_type(8)))  float    v8f;
typedef __attribute__((ext_vector_type(4)))  unsigned v4u;
typedef __attribute__((ext_vector_type(8)))  unsigned v8u;

#define B_Q    2048
#define N_C    65536
#define D_IN   256
#define DIMF   512
#define DBF    1024
#define NBLK   2
#define DOUT   10
#define BN_EPS 1e-5f
#define EPSL   1e-7f
#define SPLIT  16        // candidate-dimension split for the distance kernels

__device__ __forceinline__ v16h cat8(v8h lo, v8h hi) {
  v16h r;
#pragma unroll
  for (int i = 0; i < 8; ++i) { r[i] = lo[i]; r[i + 8] = hi[i]; }
  return r;
}

__device__ __forceinline__ v8f wmma_f16(v16h a, v16h b, v8f c) {
  // D = A(16x32 f16) * B(32x16 f16) + C(16x16 f32)
  return __builtin_amdgcn_wmma_f32_16x16x32_f16(false, a, false, b,
                                                (short)0, c, false, false);
}

// Tensor Data Mover: DMA a [rows x cols] f16 tile (row stride = rowStride
// elements) from global memory into LDS at lds_off. Issue from ONE wave only;
// follow with s_wait_tensorcnt + barrier before consumers read LDS.
__device__ __forceinline__ void tdm_load_tile_f16(unsigned lds_off,
                                                  const _Float16* gsrc,
                                                  unsigned rows, unsigned cols,
                                                  unsigned rowStride) {
  unsigned long long ga = (unsigned long long)(uintptr_t)gsrc;
  v4u g0;
  g0[0] = 1u;                                    // count=1, user descriptor
  g0[1] = lds_off;                               // lds_addr (bytes)
  g0[2] = (unsigned)(ga & 0xFFFFFFFFu);          // global_addr[31:0]
  g0[3] = (unsigned)((ga >> 32) & 0x1FFFFFFu)    // global_addr[56:32]
          | (2u << 30);                          // type = 2 ("image")
  v8u g1;
  g1[0] = 1u << 16;                              // data_size=1 (2 bytes/elem)
  g1[1] = (cols & 0xFFFFu) << 16;                // tensor_dim0[15:0] @ bits 63:48
  g1[2] = ((cols >> 16) & 0xFFFFu)               // tensor_dim0[31:16]
          | ((rows & 0xFFFFu) << 16);            // tensor_dim1[15:0]
  g1[3] = ((rows >> 16) & 0xFFFFu)               // tensor_dim1[31:16]
          | ((cols & 0xFFFFu) << 16);            // tile_dim0
  g1[4] = rows & 0xFFFFu;                        // tile_dim1 (tile_dim2 = 0)
  g1[5] = rowStride;                             // tensor_dim0_stride[31:0]
  g1[6] = 0u;                                    // stride hi / dim1_stride
  g1[7] = 0u;
  asm volatile("tensor_load_to_lds %0, %1" :: "s"(g0), "s"(g1) : "memory");
}

// ---------------------------- prep kernels ---------------------------------

__global__ void k_cvt_f16(const float* __restrict__ in,
                          _Float16* __restrict__ out, int n) {
  int i = blockIdx.x * blockDim.x + threadIdx.x;
  if (i < n) out[i] = (_Float16)in[i];
}

// W f32 [K][N] row-major  ->  Wt f16 [N][K] (so WMMA B-frags are contiguous)
__global__ void k_transpose_f16(const float* __restrict__ W,
                                _Float16* __restrict__ Wt, int K, int N) {
  int i = blockIdx.x * blockDim.x + threadIdx.x;   // i = n*K + k
  if (i < K * N) {
    int n = i / K;
    int k = i - n * K;
    Wt[i] = (_Float16)W[(size_t)k * N + n];
  }
}

// YhT[16][N] f16: row d<10 -> (y[j]==d), row 10 -> 1 (denominator), 11..15 -> 0
__global__ void k_build_yht(const int* __restrict__ y,
                            _Float16* __restrict__ YhT) {
  int j = blockIdx.x * blockDim.x + threadIdx.x;
  if (j < N_C) {
    int lab = y[j];
#pragma unroll
    for (int d = 0; d < 16; ++d) {
      float v = (d == 10) ? 1.f : ((d == lab) ? 1.f : 0.f);
      YhT[(size_t)d * N_C + j] = (_Float16)v;
    }
  }
}

// ------------------------------ WMMA GEMM ----------------------------------
// Out[M][N] f16 = BN?(relu?(A[M][K] f16 @ W[K][N] + bias)), W transposed f16.
// Block = 256 threads = 8 waves (4 in M x 2 in N); wave tile 32x64 (8 accs).
// K-loop is ping-pong software-pipelined (two disjoint fragment register
// sets) so each WMMA group reads directly from its load destinations:
// no inter-iteration register copies, loads fully overlap the XDL work.
__device__ __forceinline__ void load_frags(
    const _Float16* a0r, const _Float16* a1r,
    const _Float16* b0r, const _Float16* b1r,
    const _Float16* b2r, const _Float16* b3r, int k,
    v16h& a0, v16h& a1, v16h& b0, v16h& b1, v16h& b2, v16h& b3) {
  a0 = cat8(*(const v8h*)(a0r + k), *(const v8h*)(a0r + k + 8));
  a1 = cat8(*(const v8h*)(a1r + k), *(const v8h*)(a1r + k + 8));
  b0 = *(const v16h*)(b0r + k);
  b1 = *(const v16h*)(b1r + k);
  b2 = *(const v16h*)(b2r + k);
  b3 = *(const v16h*)(b3r + k);
}

__device__ __forceinline__ void wmma8(v8f (&acc)[2][4], v16h a0, v16h a1,
                                      v16h b0, v16h b1, v16h b2, v16h b3) {
  acc[0][0] = wmma_f16(a0, b0, acc[0][0]);
  acc[0][1] = wmma_f16(a0, b1, acc[0][1]);
  acc[0][2] = wmma_f16(a0, b2, acc[0][2]);
  acc[0][3] = wmma_f16(a0, b3, acc[0][3]);
  acc[1][0] = wmma_f16(a1, b0, acc[1][0]);
  acc[1][1] = wmma_f16(a1, b1, acc[1][1]);
  acc[1][2] = wmma_f16(a1, b2, acc[1][2]);
  acc[1][3] = wmma_f16(a1, b3, acc[1][3]);
}

__global__ __launch_bounds__(256)
void k_gemm_wmma(const _Float16* __restrict__ A, const _Float16* __restrict__ Wt,
                 const float* __restrict__ bias,
                 const float* __restrict__ bn_g, const float* __restrict__ bn_b,
                 const float* __restrict__ bn_m, const float* __restrict__ bn_v,
                 _Float16* __restrict__ Out,
                 int M, int N, int K, int relu) {   // requires K % 64 == 0
  const int lane   = threadIdx.x & 31;
  const int wave   = threadIdx.x >> 5;
  const int laneLo = lane & 15;
  const int laneHi = lane >> 4;
  const int wm = wave & 3, wn = wave >> 2;
  const int m0 = blockIdx.y * 128 + wm * 32;
  const int n0 = blockIdx.x * 128 + wn * 64;

  const _Float16* a0r = A  + (size_t)(m0 +  0 + laneLo) * K + 16 * laneHi;
  const _Float16* a1r = A  + (size_t)(m0 + 16 + laneLo) * K + 16 * laneHi;
  const _Float16* b0r = Wt + (size_t)(n0 +  0 + laneLo) * K + 16 * laneHi;
  const _Float16* b1r = Wt + (size_t)(n0 + 16 + laneLo) * K + 16 * laneHi;
  const _Float16* b2r = Wt + (size_t)(n0 + 32 + laneLo) * K + 16 * laneHi;
  const _Float16* b3r = Wt + (size_t)(n0 + 48 + laneLo) * K + 16 * laneHi;

  v8f acc[2][4];
#pragma unroll
  for (int u = 0; u < 2; ++u)
#pragma unroll
    for (int t = 0; t < 4; ++t) acc[u][t] = (v8f){};

  v16h a0A, a1A, b0A, b1A, b2A, b3A;      // ping
  v16h a0B, a1B, b0B, b1B, b2B, b3B;      // pong
  load_frags(a0r, a1r, b0r, b1r, b2r, b3r, 0, a0A, a1A, b0A, b1A, b2A, b3A);

  for (int k0 = 0; k0 < K; k0 += 64) {
    __builtin_prefetch((const void*)(a0r + k0 + 64), 0, 1);
    load_frags(a0r, a1r, b0r, b1r, b2r, b3r, k0 + 32,
               a0B, a1B, b0B, b1B, b2B, b3B);
    wmma8(acc, a0A, a1A, b0A, b1A, b2A, b3A);
    if (k0 + 64 < K)
      load_frags(a0r, a1r, b0r, b1r, b2r, b3r, k0 + 64,
                 a0A, a1A, b0A, b1A, b2A, b3A);
    wmma8(acc, a0B, a1B, b0B, b1B, b2B, b3B);
  }

  const bool hasBN = (bn_g != nullptr);
#pragma unroll
  for (int t = 0; t < 4; ++t) {
    const int n = n0 + t * 16 + laneLo;
    const float bs = bias[n];
    float sc = 1.f, sh = 0.f;
    if (hasBN) {      // fold next layer's eval-BN into this epilogue
      const float is = bn_g[n] * rsqrtf(bn_v[n] + BN_EPS);
      sc = is;
      sh = bn_b[n] - bn_m[n] * is;
    }
#pragma unroll
    for (int u = 0; u < 2; ++u) {
#pragma unroll
      for (int r = 0; r < 8; ++r) {
        float vv = acc[u][t][r] + bs;
        if (relu)  vv = fmaxf(vv, 0.f);
        if (hasBN) vv = vv * sc + sh;
        Out[(size_t)(m0 + u * 16 + 8 * laneHi + r) * N + n] = (_Float16)vv;
      }
    }
  }
}

// final BN + row sum-of-squares (Gram trick for cdist)
__global__ __launch_bounds__(128)
void k_fin_bn_norm(const _Float16* __restrict__ z, _Float16* __restrict__ e,
                   float* __restrict__ rn,
                   const float* __restrict__ g, const float* __restrict__ b,
                   const float* __restrict__ m, const float* __restrict__ v) {
  const int row = blockIdx.x;
  const int t   = threadIdx.x;
  const _Float16* zr = z + (size_t)row * DIMF;
  _Float16*       er = e + (size_t)row * DIMF;
  float ss = 0.f;
  for (int c = t; c < DIMF; c += 128) {
    float s   = g[c] * rsqrtf(v[c] + BN_EPS);
    float val = s * ((float)zr[c] - m[c]) + b[c];
    er[c] = (_Float16)val;
    ss += val * val;
  }
#pragma unroll
  for (int off = 16; off; off >>= 1) ss += __shfl_xor(ss, off, 32);
  __shared__ float ws4[4];
  if ((t & 31) == 0) ws4[t >> 5] = ss;
  __syncthreads();
  if (t == 0) rn[row] = ws4[0] + ws4[1] + ws4[2] + ws4[3];
}

// --------------------- distance pass 1: per-row min d^2 --------------------
__global__ __launch_bounds__(256)
void k_d2min(const _Float16* __restrict__ Eq, const _Float16* __restrict__ Ec,
             const float* __restrict__ qn, const float* __restrict__ cn,
             unsigned int* __restrict__ d2min) {
  __shared__ __align__(32) _Float16 Qs[16 * DIMF];
  __shared__ float qns[16];
  const int q0      = blockIdx.y * 16;
  const int colBase = blockIdx.x * (N_C / SPLIT);
  const int tid     = threadIdx.x;
  const int lane = tid & 31, wave = tid >> 5;
  const int laneLo = lane & 15, laneHi = lane >> 4;

  if (wave == 0) {    // TDM stages the 16x512 f16 Q tile into LDS
    tdm_load_tile_f16((unsigned)(uintptr_t)&Qs[0],
                      Eq + (size_t)q0 * DIMF, 16u, (unsigned)DIMF, (unsigned)DIMF);
    __builtin_amdgcn_s_wait_tensorcnt(0);
  }
  if (tid < 16) qns[tid] = qn[q0 + tid];
  __syncthreads();

  float qr[8], dmin[8];
#pragma unroll
  for (int r = 0; r < 8; ++r) { qr[r] = qns[8 * laneHi + r]; dmin[r] = 3.0e38f; }

  const int nChunks = (N_C / SPLIT) / 32;   // 128 -> 16 per wave
  for (int c = wave; c < nChunks; c += 8) {
    const int jb = colBase + c * 32;
    const _Float16* c0p = Ec + (size_t)(jb + laneLo) * DIMF + 16 * laneHi;
    const _Float16* c1p = Ec + (size_t)(jb + 16 + laneLo) * DIMF + 16 * laneHi;
    __builtin_prefetch((const void*)(c0p + 256 * DIMF), 0, 1);   // wave's next chunk
    v8f s0 = {}, s1 = {};
    v16h nb0 = *(const v16h*)(c0p), nb1 = *(const v16h*)(c1p);
#pragma unroll
    for (int k0 = 0; k0 < DIMF; k0 += 32) {
      v16h u0 = nb0, u1 = nb1;
      if (k0 + 32 < DIMF) {
        nb0 = *(const v16h*)(c0p + k0 + 32);
        nb1 = *(const v16h*)(c1p + k0 + 32);
      }
      v8h alo = *(const v8h*)(Qs + laneLo * DIMF + k0 + 16 * laneHi);
      v8h ahi = *(const v8h*)(Qs + laneLo * DIMF + k0 + 8 + 16 * laneHi);
      v16h a  = cat8(alo, ahi);
      s0 = wmma_f16(a, u0, s0);
      s1 = wmma_f16(a, u1, s1);
    }
    float cn0 = cn[jb + laneLo], cn1 = cn[jb + 16 + laneLo];
#pragma unroll
    for (int r = 0; r < 8; ++r) {
      float d20 = fmaxf(qr[r] + cn0 - 2.f * s0[r], 1e-30f);
      float d21 = fmaxf(qr[r] + cn1 - 2.f * s1[r], 1e-30f);
      dmin[r] = fminf(dmin[r], fminf(d20, d21));
    }
  }
#pragma unroll
  for (int r = 0; r < 8; ++r) {
#pragma unroll
    for (int off = 1; off < 16; off <<= 1)
      dmin[r] = fminf(dmin[r], __shfl_xor(dmin[r], off, 32));
    if (laneLo == 0)   // non-negative floats: uint ordering == float ordering
      atomicMin(&d2min[q0 + 8 * laneHi + r], __float_as_uint(dmin[r]));
  }
}

// ----- distance pass 2: p = exp(dmin - dist); class bins via WMMA vs YhT ---
__global__ __launch_bounds__(256)
void k_nca_accum(const _Float16* __restrict__ Eq, const _Float16* __restrict__ Ec,
                 const float* __restrict__ qn, const float* __restrict__ cn,
                 const unsigned int* __restrict__ d2min,
                 const _Float16* __restrict__ YhT,
                 float* __restrict__ numer_part) {
  __shared__ __align__(32) _Float16 Qs[16 * DIMF];
  __shared__ __align__(32) _Float16 Pbuf[8][16 * 32];
  __shared__ float WaveNum[8][16][16];
  __shared__ float qns[16], rms[16];
  const int q0      = blockIdx.y * 16;
  const int colBase = blockIdx.x * (N_C / SPLIT);
  const int tid     = threadIdx.x;
  const int lane = tid & 31, wave = tid >> 5;
  const int laneLo = lane & 15, laneHi = lane >> 4;

  if (wave == 0) {    // TDM stages the 16x512 f16 Q tile into LDS
    tdm_load_tile_f16((unsigned)(uintptr_t)&Qs[0],
                      Eq + (size_t)q0 * DIMF, 16u, (unsigned)DIMF, (unsigned)DIMF);
    __builtin_amdgcn_s_wait_tensorcnt(0);
  }
  if (tid < 16) {
    qns[tid] = qn[q0 + tid];
    rms[tid] = sqrtf(fmaxf(__uint_as_float(d2min[q0 + tid]), 1e-30f));
  }
  __syncthreads();

  float qr[8], rm[8];
#pragma unroll
  for (int r = 0; r < 8; ++r) {
    qr[r] = qns[8 * laneHi + r];
    rm[r] = rms[8 * laneHi + r];
  }

  v8f num = {};                           // 16 queries x 16 bins accumulator
  const int nChunks = (N_C / SPLIT) / 32; // every wave runs exactly 16 iters
  for (int c = wave; c < nChunks; c += 8) {
    const int jb = colBase + c * 32;
    const _Float16* c0p = Ec + (size_t)(jb + laneLo) * DIMF + 16 * laneHi;
    const _Float16* c1p = Ec + (size_t)(jb + 16 + laneLo) * DIMF + 16 * laneHi;
    __builtin_prefetch((const void*)(c0p + 256 * DIMF), 0, 1);   // wave's next chunk
    v8f s0 = {}, s1 = {};
    v16h nb0 = *(const v16h*)(c0p), nb1 = *(const v16h*)(c1p);
#pragma unroll
    for (int k0 = 0; k0 < DIMF; k0 += 32) {
      v16h u0 = nb0, u1 = nb1;
      if (k0 + 32 < DIMF) {
        nb0 = *(const v16h*)(c0p + k0 + 32);
        nb1 = *(const v16h*)(c1p + k0 + 32);
      }
      v8h alo = *(const v8h*)(Qs + laneLo * DIMF + k0 + 16 * laneHi);
      v8h ahi = *(const v8h*)(Qs + laneLo * DIMF + k0 + 8 + 16 * laneHi);
      v16h a  = cat8(alo, ahi);
      s0 = wmma_f16(a, u0, s0);
      s1 = wmma_f16(a, u1, s1);
    }
    float cn0 = cn[jb + laneLo], cn1 = cn[jb + 16 + laneLo];
#pragma unroll
    for (int r = 0; r < 8; ++r) {
      float d0 = sqrtf(fmaxf(qr[r] + cn0 - 2.f * s0[r], 1e-30f));
      float d1 = sqrtf(fmaxf(qr[r] + cn1 - 2.f * s1[r], 1e-30f));
      float p0 = __expf(rm[r] - d0);     // <= 1, f16-safe
      float p1 = __expf(rm[r] - d1);
      const int mm = r + 8 * laneHi;
      Pbuf[wave][mm * 32 + laneLo]      = (_Float16)p0;
      Pbuf[wave][mm * 32 + 16 + laneLo] = (_Float16)p1;
    }
    __syncthreads();   // uniform: every wave executes the same 16 iterations
    // transpose P tile (D-layout -> A-fragment layout) via LDS
    v8h plo = *(const v8h*)(&Pbuf[wave][laneLo * 32 + 16 * laneHi]);
    v8h phi = *(const v8h*)(&Pbuf[wave][laneLo * 32 + 8 + 16 * laneHi]);
    v16h ap = cat8(plo, phi);
    v16h by = *(const v16h*)(YhT + (size_t)laneLo * N_C + jb + 16 * laneHi);
    num = wmma_f16(ap, by, num);         // bins 0..9 + denominator in col 10
  }

  // deterministic per-block combine of the 8 wave tiles
#pragma unroll
  for (int r = 0; r < 8; ++r)
    WaveNum[wave][r + 8 * laneHi][laneLo] = num[r];
  __syncthreads();
  {
    const int mm = tid >> 4, d = tid & 15;
    float s = 0.f;
#pragma unroll
    for (int w = 0; w < 8; ++w) s += WaveNum[w][mm][d];
    numer_part[((size_t)blockIdx.x * B_Q + (q0 + mm)) * 16 + d] = s;
  }
}

__global__ void k_final_out(const float* __restrict__ numer_part,
                            float* __restrict__ out) {
  int i = blockIdx.x * blockDim.x + threadIdx.x;
  if (i >= B_Q * DOUT) return;
  int b = i / DOUT, d = i - b * DOUT;
  float nu = 0.f, de = 0.f;
  for (int s = 0; s < SPLIT; ++s) {
    const float* p = numer_part + ((size_t)s * B_Q + b) * 16;
    nu += p[d];
    de += p[10];
  }
  out[i] = logf(nu / fmaxf(de, 1e-30f) + EPSL);
}

// ------------------------------- launcher ----------------------------------

extern "C" void kernel_launch(void* const* d_in, const int* in_sizes, int n_in,
                              void* d_out, int out_size, void* d_ws, size_t ws_size,
                              hipStream_t stream) {
  (void)in_sizes; (void)n_in; (void)out_size; (void)ws_size;
  const float* x     = (const float*)d_in[0];
  const float* cx    = (const float*)d_in[2];
  const int*   cy    = (const int*)d_in[3];
  const float* encW  = (const float*)d_in[5];
  const float* encB  = (const float*)d_in[6];
  const float* blkW1 = (const float*)d_in[7];
  const float* blkB1 = (const float*)d_in[8];
  const float* blkW2 = (const float*)d_in[9];
  const float* blkB2 = (const float*)d_in[10];
  const float* bn_g  = (const float*)d_in[11];
  const float* bn_b  = (const float*)d_in[12];
  const float* bn_m  = (const float*)d_in[13];
  const float* bn_v  = (const float*)d_in[14];
  const float* fin_g = (const float*)d_in[15];
  const float* fin_b = (const float*)d_in[16];
  const float* fin_m = (const float*)d_in[17];
  const float* fin_v = (const float*)d_in[18];
  float* out = (float*)d_out;

  char* wp = (char*)d_ws;
  auto alloc = [&](size_t bytes) -> void* {
    void* r = (void*)wp;
    wp += (bytes + 255) & ~(size_t)255;
    return r;
  };
  _Float16* xh    = (_Float16*)alloc((size_t)B_Q * D_IN * 2);
  _Float16* cxh   = (_Float16*)alloc((size_t)N_C * D_IN * 2);
  _Float16* encWT = (_Float16*)alloc((size_t)DIMF * D_IN * 2);
  _Float16* W1T   = (_Float16*)alloc((size_t)NBLK * DBF * DIMF * 2);
  _Float16* W2T   = (_Float16*)alloc((size_t)NBLK * DIMF * DBF * 2);
  _Float16* Zq    = (_Float16*)alloc((size_t)B_Q * DIMF * 2);
  _Float16* Eq    = (_Float16*)alloc((size_t)B_Q * DIMF * 2);
  _Float16* Hq    = (_Float16*)alloc((size_t)B_Q * DBF * 2);
  _Float16* Zc    = (_Float16*)alloc((size_t)N_C * DIMF * 2);
  _Float16* Ec    = (_Float16*)alloc((size_t)N_C * DIMF * 2);
  _Float16* Hc    = (_Float16*)alloc((size_t)N_C * DBF * 2);
  float*        qnrm  = (float*)alloc((size_t)B_Q * 4);
  float*        cnrm  = (float*)alloc((size_t)N_C * 4);
  unsigned int* d2min = (unsigned int*)alloc((size_t)B_Q * 4);
  _Float16*     YhT   = (_Float16*)alloc((size_t)16 * N_C * 2);
  float* numer_part   = (float*)alloc((size_t)SPLIT * B_Q * 16 * 4);

  // 0x7F7F7F7F == 3.39e38 as float; valid +inf-ish init for uint-ordered min
  hipMemsetAsync(d2min, 0x7F, (size_t)B_Q * 4, stream);

  k_cvt_f16<<<(B_Q * D_IN + 255) / 256, 256, 0, stream>>>(x, xh, B_Q * D_IN);
  k_cvt_f16<<<(N_C * D_IN + 255) / 256, 256, 0, stream>>>(cx, cxh, N_C * D_IN);
  k_transpose_f16<<<(D_IN * DIMF + 255) / 256, 256, 0, stream>>>(encW, encWT, D_IN, DIMF);
  for (int i = 0; i < NBLK; ++i) {
    k_transpose_f16<<<(DIMF * DBF + 255) / 256, 256, 0, stream>>>(
        blkW1 + (size_t)i * DIMF * DBF, W1T + (size_t)i * DBF * DIMF, DIMF, DBF);
    k_transpose_f16<<<(DBF * DIMF + 255) / 256, 256, 0, stream>>>(
        blkW2 + (size_t)i * DBF * DIMF, W2T + (size_t)i * DIMF * DBF, DBF, DIMF);
  }
  k_build_yht<<<(N_C + 255) / 256, 256, 0, stream>>>(cy, YhT);

  // encode: GEMMs with fused bias/ReLU/next-layer-BN epilogues
  auto encode = [&](const _Float16* inh, _Float16* Z, _Float16* E, _Float16* H,
                    float* rn, int M) {
    // t0 = BN0(x @ encW + encB)        -> E
    dim3 g1(DIMF / 128, M / 128);
    k_gemm_wmma<<<g1, 256, 0, stream>>>(inh, encWT, encB,
                                        bn_g, bn_b, bn_m, bn_v, E,
                                        M, DIMF, D_IN, 0);
    for (int i = 0; i < NBLK; ++i) {
      // h = relu(t @ W1_i + b1_i)      -> H
      dim3 g2(DBF / 128, M / 128);
      k_gemm_wmma<<<g2, 256, 0, stream>>>(E, W1T + (size_t)i * DBF * DIMF,
                                          blkB1 + i * DBF,
                                          nullptr, nullptr, nullptr, nullptr, H,
                                          M, DBF, DIMF, 1);
      // z = h @ W2_i + b2_i, fused with BN_{i+1} (or raw for last block)
      const bool last = (i == NBLK - 1);
      const float* g = last ? nullptr : bn_g + (i + 1) * DIMF;
      const float* b = last ? nullptr : bn_b + (i + 1) * DIMF;
      const float* m = last ? nullptr : bn_m + (i + 1) * DIMF;
      const float* v = last ? nullptr : bn_v + (i + 1) * DIMF;
      dim3 g3(DIMF / 128, M / 128);
      k_gemm_wmma<<<g3, 256, 0, stream>>>(H, W2T + (size_t)i * DIMF * DBF,
                                          blkB2 + i * DIMF, g, b, m, v,
                                          last ? Z : E,
                                          M, DIMF, DBF, 0);
    }
    // E = BN_fin(Z), row sum-of-squares
    k_fin_bn_norm<<<M, 128, 0, stream>>>(Z, E, rn, fin_g, fin_b, fin_m, fin_v);
  };
  encode(xh, Zq, Eq, Hq, qnrm, B_Q);
  encode(cxh, Zc, Ec, Hc, cnrm, N_C);

  dim3 gd(SPLIT, B_Q / 16);
  k_d2min<<<gd, 256, 0, stream>>>(Eq, Ec, qnrm, cnrm, d2min);
  k_nca_accum<<<gd, 256, 0, stream>>>(Eq, Ec, qnrm, cnrm, d2min, YhT, numer_part);
  k_final_out<<<(B_Q * DOUT + 255) / 256, 256, 0, stream>>>(numer_part, out);
}
